// MultiHeadAttention_73315091743077
// MI455X (gfx1250) — matrix operand order repaired and checked by reading, not compiled, not run on hardware
//
#include <hip/hip_runtime.h>
#include <hip/hip_bf16.h>

#define S_LEN 2048
#define BATCH 2
#define DMODEL 1024
#define NHEAD 16
#define DK 64
#define ROWS (S_LEN * BATCH) /* 4096 */

typedef __attribute__((ext_vector_type(16))) __bf16 bf16x16;
typedef __attribute__((ext_vector_type(8)))  __bf16 bf16x8;
typedef __attribute__((ext_vector_type(8)))  float  f32x8;

union Frag16 { bf16x16 v; bf16x8 h[2]; };

__device__ __forceinline__ f32x8 wmma_bf16(bf16x16 a, bf16x16 b, f32x8 c) {
  return __builtin_amdgcn_wmma_f32_16x16x32_bf16(false, a, false, b, (short)0, c,
                                                 false, false);
}

// A-matrix 16x32 bf16 fragment (ISA 7.12.2): lane holds row m=lane&15;
// elems 0..7 -> K = k0 + 8*hi + e ; elems 8..15 -> K = k0 + 16 + 8*hi + (e-8)
__device__ __forceinline__ bf16x16 load_fragA(const __bf16* base, int ld,
                                              int row0, int k0, int lane) {
  const int m  = row0 + (lane & 15);
  const int hs = (lane >> 4) & 1;
  const __bf16* p = base + (size_t)m * ld + k0 + hs * 8;
  Frag16 f;
  f.h[0] = *(const bf16x8*)(p);
  f.h[1] = *(const bf16x8*)(p + 16);
  return f.v;
}

// B-matrix 32x16 bf16 fragment: lane holds col n=lane&15;
// elems 0..7 -> K = k0 + 8*hi + e ; elems 8..15 -> K = k0 + 16 + 8*hi + (e-8).
// Source stored N-major with K contiguous (i.e. "B^T row-major"), ld = row pitch.
__device__ __forceinline__ bf16x16 load_fragB(const __bf16* base, int ld,
                                              int n0, int k0, int lane) {
  const int n  = n0 + (lane & 15);
  const int hs = (lane >> 4) & 1;
  const __bf16* p = base + (size_t)n * ld + k0 + hs * 8;
  Frag16 f;
  f.h[0] = *(const bf16x8*)(p);
  f.h[1] = *(const bf16x8*)(p + 16);
  return f.v;
}

// 16-lane butterfly reductions in pure VALU via DPP16 (no LDS traffic).
// quad_perm(xor1)=0xB1, quad_perm(xor2)=0x4E, row_half_mirror=0x141,
// row_mirror=0x140.  Never crosses the 16-lane group boundary.
#define DPPF(x, ctrl)                                             \
  __builtin_bit_cast(float, __builtin_amdgcn_update_dpp(          \
      0, __builtin_bit_cast(int, (x)), (ctrl), 0xF, 0xF, true))

__device__ __forceinline__ float redmax16(float x) {
  x = fmaxf(x, DPPF(x, 0xB1));
  x = fmaxf(x, DPPF(x, 0x4E));
  x = fmaxf(x, DPPF(x, 0x141));
  x = fmaxf(x, DPPF(x, 0x140));
  return x;
}
__device__ __forceinline__ float redsum16(float x) {
  x += DPPF(x, 0xB1);
  x += DPPF(x, 0x4E);
  x += DPPF(x, 0x141);
  x += DPPF(x, 0x140);
  return x;
}

// ---------------------------------------------------------------- convert
__global__ __launch_bounds__(256) void cvt_f32_to_bf16(
    const float* __restrict__ in, __bf16* __restrict__ out, int n) {
  const int i = (blockIdx.x * 256 + threadIdx.x) * 4;
  if (i + 3 < n) {
    const float4 v = *(const float4*)(in + i);
    out[i + 0] = (__bf16)v.x;
    out[i + 1] = (__bf16)v.y;
    out[i + 2] = (__bf16)v.z;
    out[i + 3] = (__bf16)v.w;
  }
}

// ------------------------------------------------- NT GEMM: Y = X * W^T + b
// X:[4096,1024] bf16 row-major (rows = s*B+b), W:[1024,1024] bf16 row-major.
// MODE 0: bf16 output scattered into [B,H,S,DK];  MODE 1: f32 row-major out.
template <int MODE>
__global__ __launch_bounds__(256) void gemm_nt_kernel(
    const __bf16* __restrict__ X, const __bf16* __restrict__ W,
    const float* __restrict__ bias, void* __restrict__ outp) {
  const int wave = threadIdx.x >> 5;
  const int lane = threadIdx.x & 31;
  const int wm = wave >> 1, wn = wave & 1;
  const int m0 = blockIdx.x * 128 + wm * 32;
  const int n0 = blockIdx.y * 64 + wn * 32;

  f32x8 acc[2][2];
#pragma unroll
  for (int i = 0; i < 2; ++i)
#pragma unroll
    for (int j = 0; j < 2; ++j)
      acc[i][j] = (f32x8){0.f, 0.f, 0.f, 0.f, 0.f, 0.f, 0.f, 0.f};

  for (int k0 = 0; k0 < DMODEL; k0 += 32) {
    const bf16x16 a0 = load_fragA(X, DMODEL, m0, k0, lane);
    const bf16x16 a1 = load_fragA(X, DMODEL, m0 + 16, k0, lane);
    const bf16x16 b0 = load_fragB(W, DMODEL, n0, k0, lane);
    const bf16x16 b1 = load_fragB(W, DMODEL, n0 + 16, k0, lane);
    acc[0][0] = wmma_bf16(a0, b0, acc[0][0]);
    acc[0][1] = wmma_bf16(a0, b1, acc[0][1]);
    acc[1][0] = wmma_bf16(a1, b0, acc[1][0]);
    acc[1][1] = wmma_bf16(a1, b1, acc[1][1]);
  }

  const int hi = lane >> 4, ln = lane & 15;
#pragma unroll
  for (int j = 0; j < 2; ++j) {
    const int n = n0 + j * 16 + ln;
    const float bv = bias[n];
#pragma unroll
    for (int i = 0; i < 2; ++i) {
#pragma unroll
      for (int r = 0; r < 8; ++r) {
        const int m = m0 + i * 16 + r + 8 * hi;
        const float val = acc[i][j][r] + bv;
        if (MODE == 0) {
          const int s = m >> 1, b = m & 1;
          const int h = n >> 6, dk = n & 63;
          ((__bf16*)outp)[((size_t)(b * NHEAD + h) * S_LEN + s) * DK + dk] =
              (__bf16)val;
        } else {
          ((float*)outp)[(size_t)m * DMODEL + n] = val;
        }
      }
    }
  }
}

// ------------------------------------------------- flash attention
// grid = (S/128, B*H); block = 256 (8 waves); wave owns 16 query rows.
__global__ __launch_bounds__(256) void attn_kernel(
    const __bf16* __restrict__ Qp, const __bf16* __restrict__ Kp,
    const __bf16* __restrict__ Vp, __bf16* __restrict__ ctx) {
  __shared__ __align__(16) __bf16 ldsK[32 * DK];       // [kv][dk]
  __shared__ __align__(16) __bf16 ldsVt[DK * 32];      // [dk][kv] (transposed)
  __shared__ __align__(16) __bf16 ldsP[8][16 * 32];    // per-wave P staging

  const int wave = threadIdx.x >> 5;
  const int lane = threadIdx.x & 31;
  const int hi = lane >> 4, ln = lane & 15;
  const int bh = blockIdx.y;
  const int b = bh >> 4, h = bh & 15;
  const size_t base = (size_t)bh * S_LEN * DK;
  const __bf16* Qb = Qp + base;
  const __bf16* Kb = Kp + base;
  const __bf16* Vb = Vp + base;
  const int q0 = blockIdx.x * 128 + wave * 16;

  const bf16x16 aq0 = load_fragA(Qb, DK, q0, 0, lane);
  const bf16x16 aq1 = load_fragA(Qb, DK, q0, 32, lane);

  f32x8 o[4];
#pragma unroll
  for (int j = 0; j < 4; ++j)
    o[j] = (f32x8){0.f, 0.f, 0.f, 0.f, 0.f, 0.f, 0.f, 0.f};
  float mrow[8], lrow[8];
#pragma unroll
  for (int r = 0; r < 8; ++r) { mrow[r] = -1e30f; lrow[r] = 0.f; }

  const float scale = 0.125f;  // 1/sqrt(64)

  // per-thread fixed staging slots (256 threads x 16B = 4KB = one 32x64 tile)
  const int kvS = threadIdx.x >> 3;       // 0..31
  const int d8S = (threadIdx.x & 7) * 8;  // 0..56
  const unsigned ldsK_addr =
      (unsigned)(size_t)&ldsK[kvS * DK + d8S];  // workgroup-relative LDS byte addr

  for (int kv0 = 0; kv0 < S_LEN; kv0 += 32) {
    __syncthreads();  // previous-tile consumers done before overwrite
    {
      // K tile: async DMA straight into LDS (no VGPR round trip, ASYNCcnt)
      const unsigned long long kgaddr =
          (unsigned long long)&Kb[(size_t)(kv0 + kvS) * DK + d8S];
      asm volatile("global_load_async_to_lds_b128 %0, %1, off"
                   :: "v"(ldsK_addr), "v"(kgaddr)
                   : "memory");
      // V tile: load to VGPRs, store transposed [dk][kv] into LDS
      const bf16x8 vv = *(const bf16x8*)&Vb[(size_t)(kv0 + kvS) * DK + d8S];
#pragma unroll
      for (int t = 0; t < 8; ++t) ldsVt[(d8S + t) * 32 + kvS] = vv[t];
      asm volatile("s_wait_asynccnt 0x0" ::: "memory");
    }
    __syncthreads();

    // scores: S = Q (16x64) * K^T (64x32), two 16-col tiles
    f32x8 s[2];
#pragma unroll
    for (int nt = 0; nt < 2; ++nt) {
      const bf16x16 bk0 = load_fragB(ldsK, DK, nt * 16, 0, lane);
      const bf16x16 bk1 = load_fragB(ldsK, DK, nt * 16, 32, lane);
      f32x8 c = (f32x8){0.f, 0.f, 0.f, 0.f, 0.f, 0.f, 0.f, 0.f};
      c = wmma_bf16(aq0, bk0, c);
      c = wmma_bf16(aq1, bk1, c);
      s[nt] = c;
    }

    // online softmax update (row stats per C-frag register, DPP reductions)
#pragma unroll
    for (int r = 0; r < 8; ++r) {
      const float s0 = s[0][r] * scale;
      const float s1 = s[1][r] * scale;
      const float tmax = redmax16(fmaxf(s0, s1));
      const float mnew = fmaxf(mrow[r], tmax);
      const float alpha = __expf(mrow[r] - mnew);
      const float p0 = __expf(s0 - mnew);
      const float p1 = __expf(s1 - mnew);
      lrow[r] = lrow[r] * alpha + redsum16(p0 + p1);
      mrow[r] = mnew;
#pragma unroll
      for (int j = 0; j < 4; ++j) o[j][r] *= alpha;
      const int m = r + 8 * hi;
      ldsP[wave][m * 32 + ln] = (__bf16)p0;
      ldsP[wave][m * 32 + 16 + ln] = (__bf16)p1;
    }
    asm volatile("" ::: "memory");  // keep DS store->load order; HW DS in-order

    // ctx += P (16x32) * V (32x64)
    const bf16x16 ap = load_fragA(&ldsP[wave][0], 32, 0, 0, lane);
#pragma unroll
    for (int j = 0; j < 4; ++j) {
      const bf16x16 bv = load_fragB(ldsVt, 32, j * 16, 0, lane);
      o[j] = wmma_bf16(ap, bv, o[j]);
    }
  }

  // epilogue: normalize, write ctx rows (s,b) with D contiguous for final GEMM
#pragma unroll
  for (int r = 0; r < 8; ++r) {
    const int m = r + 8 * hi;
    const int srow = q0 + m;
    const float inv = 1.f / lrow[r];
#pragma unroll
    for (int j = 0; j < 4; ++j) {
      const int col = h * DK + j * 16 + ln;
      ctx[(size_t)(srow * BATCH + b) * DMODEL + col] = (__bf16)(o[j][r] * inv);
    }
  }
}

// ----------------------------------------------------------------- launch
extern "C" void kernel_launch(void* const* d_in, const int* in_sizes, int n_in,
                              void* d_out, int out_size, void* d_ws,
                              size_t ws_size, hipStream_t stream) {
  (void)in_sizes; (void)n_in; (void)out_size; (void)ws_size;
  const float* q  = (const float*)d_in[0];
  const float* k  = (const float*)d_in[1];
  const float* v  = (const float*)d_in[2];
  const float* Wq = (const float*)d_in[3];
  const float* bq = (const float*)d_in[4];
  const float* Wk = (const float*)d_in[5];
  const float* bk = (const float*)d_in[6];
  const float* Wv = (const float*)d_in[7];
  const float* bv = (const float*)d_in[8];
  const float* Wo = (const float*)d_in[9];
  const float* bo = (const float*)d_in[10];

  char* ws = (char*)d_ws;
  const size_t SZ_ACT = (size_t)ROWS * DMODEL * sizeof(__bf16);   // 8 MB
  const size_t SZ_W   = (size_t)DMODEL * DMODEL * sizeof(__bf16); // 2 MB
  __bf16* qb  = (__bf16*)(ws + 0 * SZ_ACT);
  __bf16* kb  = (__bf16*)(ws + 1 * SZ_ACT);
  __bf16* vb  = (__bf16*)(ws + 2 * SZ_ACT);
  __bf16* Wqb = (__bf16*)(ws + 3 * SZ_ACT + 0 * SZ_W);
  __bf16* Wkb = (__bf16*)(ws + 3 * SZ_ACT + 1 * SZ_W);
  __bf16* Wvb = (__bf16*)(ws + 3 * SZ_ACT + 2 * SZ_W);
  __bf16* Wob = (__bf16*)(ws + 3 * SZ_ACT + 3 * SZ_W);
  __bf16* Qp  = (__bf16*)(ws + 3 * SZ_ACT + 4 * SZ_W);
  __bf16* Kp  = (__bf16*)(ws + 4 * SZ_ACT + 4 * SZ_W);
  __bf16* Vp  = (__bf16*)(ws + 5 * SZ_ACT + 4 * SZ_W);
  __bf16* ctx = (__bf16*)(ws + 6 * SZ_ACT + 4 * SZ_W);

  const int nact = ROWS * DMODEL;    // 4,194,304
  const int nwgt = DMODEL * DMODEL;  // 1,048,576
  dim3 blk(256);
  cvt_f32_to_bf16<<<nact / 1024, blk, 0, stream>>>(q, qb, nact);
  cvt_f32_to_bf16<<<nact / 1024, blk, 0, stream>>>(k, kb, nact);
  cvt_f32_to_bf16<<<nact / 1024, blk, 0, stream>>>(v, vb, nact);
  cvt_f32_to_bf16<<<nwgt / 1024, blk, 0, stream>>>(Wq, Wqb, nwgt);
  cvt_f32_to_bf16<<<nwgt / 1024, blk, 0, stream>>>(Wk, Wkb, nwgt);
  cvt_f32_to_bf16<<<nwgt / 1024, blk, 0, stream>>>(Wv, Wvb, nwgt);
  cvt_f32_to_bf16<<<nwgt / 1024, blk, 0, stream>>>(Wo, Wob, nwgt);

  dim3 ggrid(ROWS / 128, DMODEL / 64);
  gemm_nt_kernel<0><<<ggrid, blk, 0, stream>>>(qb, Wqb, bq, Qp);
  gemm_nt_kernel<0><<<ggrid, blk, 0, stream>>>(kb, Wkb, bk, Kp);
  gemm_nt_kernel<0><<<ggrid, blk, 0, stream>>>(vb, Wvb, bv, Vp);

  dim3 agrid(S_LEN / 128, BATCH * NHEAD);
  attn_kernel<<<agrid, blk, 0, stream>>>(Qp, Kp, Vp, ctx);

  gemm_nt_kernel<1><<<ggrid, blk, 0, stream>>>(ctx, Wob, bo, d_out);
}
